// Inference_24300924961606
// MI455X (gfx1250) — matrix operand order repaired and checked by reading, not compiled, not run on hardware
//
#include <hip/hip_runtime.h>
#include <stdint.h>

// ---------------- CDNA5 WMMA types ----------------
typedef __attribute__((ext_vector_type(16))) __bf16 v16bf;
typedef __attribute__((ext_vector_type(8)))  float  v8f;

#define TOK_TILE 64
#define NTHR     512
#define NW       (NTHR / 32)
#define HDIM_    1024
#define IDIM_    512

// bf16 weight layout inside d_ws (element offsets)
constexpr size_t OFF_EW0 = 0;
constexpr size_t OFF_EW1 = OFF_EW0 + 512 * 512;
constexpr size_t OFF_EW2 = OFF_EW1 + 512 * 512;
constexpr size_t OFF_EW3 = OFF_EW2 + 512 * 512;
constexpr size_t OFF_DW0 = OFF_EW3 + 512 * 1024;
constexpr size_t OFF_DW1 = OFF_DW0 + 1024 * 512;
constexpr size_t OFF_DW2 = OFF_DW1 + 512 * 512;
constexpr size_t OFF_DW3 = OFF_DW2 + 512 * 512;

constexpr int SMEM_BYTES = 2 * TOK_TILE * HDIM_ * 2 + TOK_TILE * 4;  // 262400 B

__device__ __forceinline__ uint16_t f32_to_bf16(float f) {
  uint32_t u = __float_as_uint(f);
  u += 0x7FFFu + ((u >> 16) & 1u);      // round-to-nearest-even
  return (uint16_t)(u >> 16);
}
__device__ __forceinline__ float bf16_to_f32(uint16_t b) {
  return __uint_as_float(((uint32_t)b) << 16);
}

union FragBF { v16bf v; uint32_t u[8]; uint4 q[2]; };

// A fragment: 16x32 bf16 tile of activations (row-major LDS, width Kdim).
// Lane's 8 dwords are two contiguous 16B chunks -> two ds_load_b128.
__device__ __forceinline__ v16bf load_a_frag(const uint16_t* act, int Kdim,
                                             int r0, int k0, int lane) {
  FragBF f;
  const int row = r0 + (lane & 15);
  const char* p = reinterpret_cast<const char*>(act + row * Kdim + k0)
                + ((lane < 16) ? 0 : 16);
  f.q[0] = *reinterpret_cast<const uint4*>(p);
  f.q[1] = *reinterpret_cast<const uint4*>(p + 32);
  return f.v;
}

// B fragment: 32x16 bf16 weight tile (row-major global, width Ndim).
// Lane l holds column n0+(l&15); K in table order: k = 8h+s (s<8), 16+8h+(s-8) (s>=8).
__device__ __forceinline__ v16bf load_b_frag(const uint16_t* W, int Ndim,
                                             int k0, int n0, int lane) {
  FragBF f;
  const uint16_t* p = W + (size_t)k0 * Ndim + n0 + (lane & 15);
  const int h8 = (lane >> 4) * 8;
#pragma unroll
  for (int j = 0; j < 8; ++j) {
    const int s = 2 * j;
    const int k = (s < 8) ? (h8 + s) : (16 + h8 + s - 8);
    f.u[j] = (uint32_t)p[(size_t)k * Ndim] | ((uint32_t)p[(size_t)(k + 1) * Ndim] << 16);
  }
  return f.v;
}

#define WMMA_BF16(A, B, C) \
  __builtin_amdgcn_wmma_f32_16x16x32_bf16(false, (A), false, (B), (short)0, (C), false, false)

// MODE: 0 = relu -> bf16 LDS, 1 = linear -> bf16 LDS, 2 = linear -> f32 global
template <int MODE>
__device__ __forceinline__ void tile_epilogue(v8f acc, int r0, int n0, int lane,
                                              float bv,
                                              uint16_t* __restrict__ actOut, int Ndim,
                                              float* __restrict__ outG, long tokBase) {
  const int col = lane & 15;
  const int rb  = r0 + ((lane < 16) ? 0 : 8);
#pragma unroll
  for (int v = 0; v < 8; ++v) {
    float xv = acc[v] + bv;
    if (MODE == 0) xv = fmaxf(xv, 0.0f);
    if (MODE <= 1) actOut[(rb + v) * Ndim + n0 + col] = f32_to_bf16(xv);
    else           outG[(tokBase + rb + v) * (long)Ndim + n0 + col] = xv;
  }
}

// Each wave owns TWO adjacent 16-wide N columns and all 4 row tiles (64 tokens):
// per k-step, 4 A fragments + 2 B fragments feed 8 independent WMMA chains.
// Loops are pinned rolled so the k-indexed LDS loads cannot be hoisted/spilled.
// MUST stay inlined: outlining turns global/LDS pointers generic -> flat loads
// + addrspacecast null guards (seen in round 4).
template <int MODE>
__device__ __forceinline__ void gemm_layer(const uint16_t* __restrict__ actIn,
                                           uint16_t* __restrict__ actOut,
                                           const uint16_t* __restrict__ W,
                                           const float* __restrict__ bias,
                                           int Kdim, int Ndim,
                                           float* __restrict__ outG, long tokBase) {
  const int tid  = threadIdx.x;
  const int lane = tid & 31;
  const int wave = tid >> 5;
  const int ncols = Ndim >> 4;
#pragma unroll 1
  for (int c0 = wave * 2; c0 < ncols; c0 += NW * 2) {
    const int n0 = c0 << 4;         // first column; second is n0 + 16
    v8f a00 = {}, a10 = {}, a20 = {}, a30 = {};  // column n0,  rows 0/16/32/48
    v8f a01 = {}, a11 = {}, a21 = {}, a31 = {};  // column n0+16
#pragma unroll 1
    for (int k0 = 0; k0 < Kdim; k0 += 32) {
      if (k0 + 32 < Kdim)  // keep next K-slab of weights warm
        __builtin_prefetch(W + (size_t)(k0 + 32 + lane) * Ndim + n0, 0, 3);
      const v16bf b0 = load_b_frag(W, Ndim, k0, n0, lane);
      const v16bf b1 = load_b_frag(W, Ndim, k0, n0 + 16, lane);
      const v16bf f0 = load_a_frag(actIn, Kdim,  0, k0, lane);
      const v16bf f1 = load_a_frag(actIn, Kdim, 16, k0, lane);
      const v16bf f2 = load_a_frag(actIn, Kdim, 32, k0, lane);
      const v16bf f3 = load_a_frag(actIn, Kdim, 48, k0, lane);
      a00 = WMMA_BF16(f0, b0, a00);
      a10 = WMMA_BF16(f1, b0, a10);
      a20 = WMMA_BF16(f2, b0, a20);
      a30 = WMMA_BF16(f3, b0, a30);
      a01 = WMMA_BF16(f0, b1, a01);
      a11 = WMMA_BF16(f1, b1, a11);
      a21 = WMMA_BF16(f2, b1, a21);
      a31 = WMMA_BF16(f3, b1, a31);
    }
    const float bv0 = bias[n0 + (lane & 15)];
    const float bv1 = bias[n0 + 16 + (lane & 15)];
    tile_epilogue<MODE>(a00,  0, n0,      lane, bv0, actOut, Ndim, outG, tokBase);
    tile_epilogue<MODE>(a10, 16, n0,      lane, bv0, actOut, Ndim, outG, tokBase);
    tile_epilogue<MODE>(a20, 32, n0,      lane, bv0, actOut, Ndim, outG, tokBase);
    tile_epilogue<MODE>(a30, 48, n0,      lane, bv0, actOut, Ndim, outG, tokBase);
    tile_epilogue<MODE>(a01,  0, n0 + 16, lane, bv1, actOut, Ndim, outG, tokBase);
    tile_epilogue<MODE>(a11, 16, n0 + 16, lane, bv1, actOut, Ndim, outG, tokBase);
    tile_epilogue<MODE>(a21, 32, n0 + 16, lane, bv1, actOut, Ndim, outG, tokBase);
    tile_epilogue<MODE>(a31, 48, n0 + 16, lane, bv1, actOut, Ndim, outG, tokBase);
  }
}

__global__ __launch_bounds__(NTHR) void sae_fused(
    const float* __restrict__ x, const float* __restrict__ mprev,
    const uint16_t* __restrict__ wq,
    const float* __restrict__ be0, const float* __restrict__ be1,
    const float* __restrict__ be2, const float* __restrict__ be3,
    const float* __restrict__ bd0, const float* __restrict__ bd1,
    const float* __restrict__ bd2, const float* __restrict__ bd3,
    float* __restrict__ out) {
  extern __shared__ char smem[];
  uint16_t* bufA   = reinterpret_cast<uint16_t*>(smem);
  uint16_t* bufB   = bufA + TOK_TILE * HDIM_;
  unsigned* tieCnt = reinterpret_cast<unsigned*>(bufB + TOK_TILE * HDIM_);
  const int  tid     = threadIdx.x;
  const long tokBase = (long)blockIdx.x * TOK_TILE;

  // ---- stage x tile: fp32 global -> bf16 LDS ----
  {
    const float4* xv  = reinterpret_cast<const float4*>(x + tokBase * IDIM_);
    uint32_t*     dst = reinterpret_cast<uint32_t*>(bufA);
    const int nv = TOK_TILE * IDIM_ / 4;
    for (int i = tid; i < nv; i += NTHR) {
      const float4 f = xv[i];
      dst[i * 2]     = (uint32_t)f32_to_bf16(f.x) | ((uint32_t)f32_to_bf16(f.y) << 16);
      dst[i * 2 + 1] = (uint32_t)f32_to_bf16(f.z) | ((uint32_t)f32_to_bf16(f.w) << 16);
    }
  }
  __syncthreads();

  // ---- encoder ----
  gemm_layer<0>(bufA, bufB, wq + OFF_EW0, be0, 512, 512,  nullptr, tokBase); __syncthreads();
  gemm_layer<0>(bufB, bufA, wq + OFF_EW1, be1, 512, 512,  nullptr, tokBase); __syncthreads();
  gemm_layer<0>(bufA, bufB, wq + OFF_EW2, be2, 512, 512,  nullptr, tokBase); __syncthreads();
  gemm_layer<1>(bufB, bufA, wq + OFF_EW3, be3, 512, 1024, nullptr, tokBase); __syncthreads();

  // ---- hidden_exclude_activation: h[mask_prev>0] = 0 ----
  {
    const float4* mv = reinterpret_cast<const float4*>(mprev + tokBase * HDIM_);
    const int nv = TOK_TILE * HDIM_ / 4;
    for (int i = tid; i < nv; i += NTHR) {
      const float4 m = mv[i];
      const int e = i * 4;
      if (m.x > 0.f) bufA[e]     = 0;
      if (m.y > 0.f) bufA[e + 1] = 0;
      if (m.z > 0.f) bufA[e + 2] = 0;
      if (m.w > 0.f) bufA[e + 3] = 0;
    }
  }
  __syncthreads();

  // ---- exact top-256 of h*h per token; one wave per token ----
  {
    const int lane = tid & 31, wave = tid >> 5;
    for (int tk = wave; tk < TOK_TILE; tk += NW) {
      uint16_t* hrow = bufA + tk * HDIM_;
      uint32_t eb[32];
#pragma unroll
      for (int j = 0; j < 32; ++j) {
        const float v = bf16_to_f32(hrow[lane * 32 + j]);
        eb[j] = __float_as_uint(v * v);  // e >= 0 -> bits monotonic
      }
      // binary search: largest thr with count(e >= thr) >= 256
      uint32_t lo = 0u, hi = 0xFF800001u;
#pragma unroll 1
      for (int it = 0; it < 32 && (hi - lo) > 1u; ++it) {
        const uint32_t mid = lo + ((hi - lo) >> 1);
        int c = 0;
#pragma unroll
        for (int j = 0; j < 32; ++j) c += (eb[j] >= mid) ? 1 : 0;
#pragma unroll
        for (int off = 16; off > 0; off >>= 1) c += __shfl_xor(c, off, 32);
        if (c >= 256) lo = mid; else hi = mid;
      }
      int cgt = 0;
#pragma unroll
      for (int j = 0; j < 32; ++j) cgt += (eb[j] > lo) ? 1 : 0;
#pragma unroll
      for (int off = 16; off > 0; off >>= 1) cgt += __shfl_xor(cgt, off, 32);
      const unsigned rem = (unsigned)(256 - cgt);  // tie slots (>=1 by invariant)
      if (lane == 0) tieCnt[tk] = 0u;              // same-wave DS ops stay in order
#pragma unroll
      for (int j = 0; j < 32; ++j) {
        bool keep = eb[j] > lo;
        if (!keep && eb[j] == lo) {
          const unsigned slot = atomicAdd(&tieCnt[tk], 1u);  // ds_add_rtn_u32
          keep = slot < rem;
        }
        if (!keep) hrow[lane * 32 + j] = 0;
      }
    }
  }
  __syncthreads();

  // ---- decoder ----
  gemm_layer<0>(bufA, bufB, wq + OFF_DW0, bd0, 1024, 512, nullptr, tokBase); __syncthreads();
  gemm_layer<0>(bufB, bufA, wq + OFF_DW1, bd1, 512,  512, nullptr, tokBase); __syncthreads();
  gemm_layer<0>(bufA, bufB, wq + OFF_DW2, bd2, 512,  512, nullptr, tokBase); __syncthreads();
  gemm_layer<2>(bufB, nullptr, wq + OFF_DW3, bd3, 512, 512, out, tokBase);
}

// ---- one-time fp32 -> bf16 weight repack into workspace ----
__global__ __launch_bounds__(256) void cvt_w(const float* __restrict__ s,
                                             uint16_t* __restrict__ d, int n) {
  const int i = (blockIdx.x * 256 + threadIdx.x) * 4;
  if (i + 3 < n) {
    const float4 f = *reinterpret_cast<const float4*>(s + i);
    *reinterpret_cast<uint32_t*>(d + i) =
        (uint32_t)f32_to_bf16(f.x) | ((uint32_t)f32_to_bf16(f.y) << 16);
    *reinterpret_cast<uint32_t*>(d + i + 2) =
        (uint32_t)f32_to_bf16(f.z) | ((uint32_t)f32_to_bf16(f.w) << 16);
  }
}

extern "C" void kernel_launch(void* const* d_in, const int* in_sizes, int n_in,
                              void* d_out, int out_size, void* d_ws, size_t ws_size,
                              hipStream_t stream) {
  const float* x  = (const float*)d_in[0];
  const float* mp = (const float*)d_in[1];
  // dict order: e_w{i}, e_b{i}, d_w{i}, d_b{i} for i = 0..3
  const float* ew[4] = {(const float*)d_in[2],  (const float*)d_in[6],
                        (const float*)d_in[10], (const float*)d_in[14]};
  const float* eb[4] = {(const float*)d_in[3],  (const float*)d_in[7],
                        (const float*)d_in[11], (const float*)d_in[15]};
  const float* dw[4] = {(const float*)d_in[4],  (const float*)d_in[8],
                        (const float*)d_in[12], (const float*)d_in[16]};
  const float* db[4] = {(const float*)d_in[5],  (const float*)d_in[9],
                        (const float*)d_in[13], (const float*)d_in[17]};
  uint16_t* wq = (uint16_t*)d_ws;

  auto cvt = [&](const float* src, size_t off, int n) {
    cvt_w<<<n / 1024, 256, 0, stream>>>(src, wq + off, n);
  };
  cvt(ew[0], OFF_EW0, 512 * 512);
  cvt(ew[1], OFF_EW1, 512 * 512);
  cvt(ew[2], OFF_EW2, 512 * 512);
  cvt(ew[3], OFF_EW3, 512 * 1024);
  cvt(dw[0], OFF_DW0, 1024 * 512);
  cvt(dw[1], OFF_DW1, 512 * 512);
  cvt(dw[2], OFF_DW2, 512 * 512);
  cvt(dw[3], OFF_DW3, 512 * 512);

  (void)hipFuncSetAttribute(reinterpret_cast<const void*>(sae_fused),
                            hipFuncAttributeMaxDynamicSharedMemorySize, SMEM_BYTES);

  const int ntok = 16 * 4096;
  sae_fused<<<ntok / TOK_TILE, NTHR, SMEM_BYTES, stream>>>(
      x, mp, wq, eb[0], eb[1], eb[2], eb[3], db[0], db[1], db[2], db[3],
      (float*)d_out);
}
